// TransformerBlock_32031866093614
// MI455X (gfx1250) — matrix operand order repaired and checked by reading, not compiled, not run on hardware
//
#include <hip/hip_runtime.h>
#include <stdint.h>

// ---------------- problem constants ----------------
#define BB   4
#define SS   2048
#define DM   512
#define HH   8
#define DK   64
#define DFF  2048
#define EPSV 1e-5f

// ---------------- WMMA types (gfx1250, wave32) ----------------
typedef __attribute__((ext_vector_type(16))) __bf16 v16bf;
typedef __attribute__((ext_vector_type(8)))  float  v8f;

union Frag16 { v16bf v; uint32_t u[8]; uint16_t e[16]; };
union FragC  { v8f   v; float    f[8]; };

// ---------------- TDM descriptor types ----------------
typedef __attribute__((ext_vector_type(4))) unsigned int u32x4;
typedef __attribute__((ext_vector_type(8))) int          i32x8;
typedef __attribute__((ext_vector_type(4))) int          i32x4;

__device__ __forceinline__ uint16_t f2bf(float f) {
  uint32_t u = __float_as_uint(f);
  u += 0x7FFFu + ((u >> 16) & 1u);       // round-to-nearest-even
  return (uint16_t)(u >> 16);
}

// Issue a 2-D TENSOR_LOAD_TO_LDS: copies tile_d1 rows of tile_d0 bf16
// elements (row stride = stride0 elements) from global into contiguous LDS.
// D# layout per CDNA5 ISA (2-byte data_size,
// no padding / iteration / multicast; groups 2,3 zero => 2-D tensor).
__device__ __forceinline__ void tdm_load_2d(unsigned lds_off, const void* gptr,
                                            unsigned tensor_d0, unsigned tensor_d1,
                                            unsigned tile_d0, unsigned tile_d1,
                                            unsigned long long stride0) {
  unsigned long long ga = (unsigned long long)(uintptr_t)gptr;
  u32x4 g0;
  g0[0] = 1u;                                             // count=1, user desc
  g0[1] = lds_off;                                        // lds_addr (bytes)
  g0[2] = (unsigned)(ga & 0xFFFFFFFFu);                   // global_addr[31:0]
  g0[3] = (unsigned)((ga >> 32) & 0x01FFFFFFu) | (2u << 30);  // addr[56:32] | type=2
  i32x8 g1;
  g1[0] = (int)(1u << 16);                                // data_size=1 (2 bytes)
  g1[1] = (int)((tensor_d0 & 0xFFFFu) << 16);             // tensor_dim0[15:0] @63:48
  g1[2] = (int)((tensor_d0 >> 16) | ((tensor_d1 & 0xFFFFu) << 16)); // dim0 hi | dim1 lo
  g1[3] = (int)((tensor_d1 >> 16) | (tile_d0 << 16));     // dim1 hi | tile_dim0
  g1[4] = (int)(tile_d1 & 0xFFFFu);                       // tile_dim1 (tile_dim2=0)
  g1[5] = (int)(stride0 & 0xFFFFFFFFull);                 // tensor_dim0_stride lo32
  g1[6] = (int)((stride0 >> 32) & 0xFFFFull);             // stride hi16 (dim1_stride=0)
  g1[7] = 0;
  i32x4 z4 = {0, 0, 0, 0};
#if defined(__clang_major__) && (__clang_major__ >= 23)
  i32x8 z8 = {0, 0, 0, 0, 0, 0, 0, 0};
  __builtin_amdgcn_tensor_load_to_lds(g0, g1, z4, z4, z8, 0);
#else
  __builtin_amdgcn_tensor_load_to_lds(g0, g1, z4, z4, 0);
#endif
}

__device__ __forceinline__ unsigned lds_addr_of(const void* p) {
  // flat LDS address: addr[31:0] is the LDS byte offset (ISA aperture rule)
  return (unsigned)(uintptr_t)p;
}

// ---------------- fp32 -> bf16 convert ----------------
__global__ __launch_bounds__(256)
void cvt_f32_bf16_kernel(const float* __restrict__ in, uint16_t* __restrict__ out, int n) {
  int i = blockIdx.x * blockDim.x + threadIdx.x;
  int stride = gridDim.x * blockDim.x;
  for (; i < n; i += stride) out[i] = f2bf(in[i]);
}

// ---------------- tiled bf16 WMMA GEMM with TDM double-buffering ----------------
// C[M,N] = A[M,K](bf16) @ W[K,N](bf16) + bias, optional ReLU, write f32 and/or
// bf16. Block = 256 thr = 8 waves; tile 128x64; wave tile 32x32 = 2x2
// WMMA(16x16x32 bf16). Tiles staged by the Tensor Data Mover (wave 0 issues),
// double-buffered so DMA of tile i+1 overlaps WMMA on tile i.
template<int RELU, int WF32, int WBF16>
__global__ __launch_bounds__(256)
void gemm_bf16_kernel(const uint16_t* __restrict__ A, const uint16_t* __restrict__ W,
                      const float* __restrict__ bias,
                      float* __restrict__ Cf, uint16_t* __restrict__ Cb,
                      int N, int K) {
  __shared__ __align__(16) uint16_t sA[2][128 * 32];   // row-major M x K32
  __shared__ __align__(16) uint16_t sB[2][32 * 64];    // row-major K32 x N64

  const int tid  = threadIdx.x;
  const int lane = tid & 31;
  const int wave = tid >> 5;
  const int wm   = wave & 3;      // 4 waves along M
  const int wn   = wave >> 2;     // 2 waves along N
  const int m0   = blockIdx.y * 128;
  const int n0   = blockIdx.x * 64;
  const int mloc = lane & 15;
  const int half = lane >> 4;
  const bool issuer = (tid < 32);  // wave 0 drives the TDM
  const int ktiles = K / 32;

  FragC acc[2][2];
  for (int i = 0; i < 2; i++)
    for (int j = 0; j < 2; j++)
      for (int v = 0; v < 8; v++) acc[i][j].f[v] = 0.f;

  // prologue: DMA tile 0 into buffer 0
  if (issuer) {
    tdm_load_2d(lds_addr_of(&sA[0][0]), A + (size_t)m0 * K,
                (unsigned)K, 128u, 32u, 128u, (unsigned long long)K);
    tdm_load_2d(lds_addr_of(&sB[0][0]), W + (size_t)0 * N + n0,
                (unsigned)N, 32u, 64u, 32u, (unsigned long long)N);
  }

  for (int it = 0; it < ktiles; ++it) {
    const int cur = it & 1;
    if (issuer) {
      if (it + 1 < ktiles) {
        int kn = (it + 1) * 32;
        tdm_load_2d(lds_addr_of(&sA[cur ^ 1][0]), A + (size_t)m0 * K + kn,
                    (unsigned)K, 128u, 32u, 128u, (unsigned long long)K);
        tdm_load_2d(lds_addr_of(&sB[cur ^ 1][0]), W + (size_t)kn * N + n0,
                    (unsigned)N, 32u, 64u, 32u, (unsigned long long)N);
        __builtin_amdgcn_s_wait_tensorcnt(2);   // tile 'it' complete, next in flight
      } else {
        __builtin_amdgcn_s_wait_tensorcnt(0);
      }
    }
    __syncthreads();

    // --- A fragments (ISA A-layout: pair p -> K = 2*(p%4) + 16*(p/4) + 8*half)
    Frag16 a[2], b[2];
    for (int i = 0; i < 2; i++) {
      int r0 = wm * 32 + 16 * i + mloc;
      for (int p = 0; p < 8; p++) {
        int k = 2 * (p & 3) + 16 * (p >> 2) + 8 * half;
        a[i].u[p] = *(const uint32_t*)&sA[cur][r0 * 32 + k];
      }
    }
    // --- B fragments (same K order as A: pair p -> K = 2*(p%4) + 16*(p/4) + 8*half; n = lane%16)
    for (int j = 0; j < 2; j++) {
      int c0 = wn * 32 + 16 * j + mloc;
      for (int p = 0; p < 8; p++) {
        int k = 2 * (p & 3) + 16 * (p >> 2) + 8 * half;
        b[j].e[2 * p]     = sB[cur][k * 64 + c0];
        b[j].e[2 * p + 1] = sB[cur][(k + 1) * 64 + c0];
      }
    }
    for (int i = 0; i < 2; i++)
      for (int j = 0; j < 2; j++)
        acc[i][j].v = __builtin_amdgcn_wmma_f32_16x16x32_bf16(
            false, a[i].v, false, b[j].v, (short)0, acc[i][j].v, false, false);
    __syncthreads();   // everyone done with buffer 'cur' before it is re-filled
  }

  // --- epilogue (C-layout: VGPR v -> row v + 8*half, col = lane%16)
  for (int i = 0; i < 2; i++)
    for (int j = 0; j < 2; j++) {
      int col = n0 + wn * 32 + 16 * j + mloc;
      float bi = bias[col];
      for (int v = 0; v < 8; v++) {
        int row = m0 + wm * 32 + 16 * i + v + 8 * half;
        float val = acc[i][j].f[v] + bi;
        if (RELU) val = fmaxf(val, 0.f);
        if (WF32) Cf[(size_t)row * N + col] = val;
        if (WBF16) Cb[(size_t)row * N + col] = f2bf(val);
      }
    }
}

// ---------------- flash-style attention with TDM staging ----------------
// 1 wave per (b, h, 16-query tile). Streams 32 keys/iter; K/V tiles DMAed by
// the TDM (double-buffered). scores = Q(16x64) Kblk^T via 2 chained WMMA;
// online softmax; P(16x32 bf16 via LDS relayout) @ V(32x64) via 4 WMMA.
// mask: allowed iff (key >= query) || (query < prefix[b]); no 1/sqrt(dk).
__global__ __launch_bounds__(32)
void attn_kernel(const uint16_t* __restrict__ Q, const uint16_t* __restrict__ Km,
                 const uint16_t* __restrict__ V, const int* __restrict__ prefix,
                 uint16_t* __restrict__ O) {
  __shared__ __align__(16) uint16_t sK[2][32 * 64];   // key-rows x dk, row-major
  __shared__ __align__(16) uint16_t sV[2][32 * 64];   // key-rows x dk, row-major
  __shared__ __align__(16) uint16_t sP[16 * 32];      // probs tile, row-major

  const int lane  = threadIdx.x;
  const int mloc  = lane & 15;
  const int half  = lane >> 4;
  const int qt    = blockIdx.x;
  const int h     = blockIdx.y;
  const int bz    = blockIdx.z;
  const int qbase = qt * 16;
  const int pfx   = prefix[bz];
  const size_t headoff = (size_t)h * DK;

  // Q fragments: two K32 steps across dk=64
  Frag16 aq[2];
  for (int s = 0; s < 2; s++)
    for (int p = 0; p < 8; p++) {
      int kf = 32 * s + 2 * (p & 3) + 16 * (p >> 2) + 8 * half;
      aq[s].u[p] = *(const uint32_t*)&Q[(size_t)(bz * SS + qbase + mloc) * DM + headoff + kf];
    }

  FragC o[4];
  for (int j = 0; j < 4; j++)
    for (int i = 0; i < 8; i++) o[j].f[i] = 0.f;
  float mrow[8], lrow[8];
  for (int i = 0; i < 8; i++) { mrow[i] = -3e38f; lrow[i] = 0.f; }

  // prologue: DMA K/V tile 0
  tdm_load_2d(lds_addr_of(&sK[0][0]), Km + (size_t)(bz * SS) * DM + headoff,
              64u, 32u, 64u, 32u, (unsigned long long)DM);
  tdm_load_2d(lds_addr_of(&sV[0][0]), V + (size_t)(bz * SS) * DM + headoff,
              64u, 32u, 64u, 32u, (unsigned long long)DM);

  const int ntiles = SS / 32;
  for (int it = 0; it < ntiles; ++it) {
    const int kb  = it * 32;
    const int cur = it & 1;
    if (it + 1 < ntiles) {
      const size_t nb = (size_t)(bz * SS + kb + 32) * DM + headoff;
      tdm_load_2d(lds_addr_of(&sK[cur ^ 1][0]), Km + nb, 64u, 32u, 64u, 32u,
                  (unsigned long long)DM);
      tdm_load_2d(lds_addr_of(&sV[cur ^ 1][0]), V + nb, 64u, 32u, 64u, 32u,
                  (unsigned long long)DM);
      __builtin_amdgcn_s_wait_tensorcnt(2);
    } else {
      __builtin_amdgcn_s_wait_tensorcnt(0);
    }
    __syncthreads();   // ordering fence (single wave)

    // scores for two 16-key subtiles
    FragC s[2];
    for (int t = 0; t < 2; t++) {
      Frag16 bk0, bk1;
      int keyl = 16 * t + mloc;
      for (int p = 0; p < 8; p++) {
        int kf = 2 * (p & 3) + 16 * (p >> 2) + 8 * half;
        bk0.u[p] = *(const uint32_t*)&sK[cur][keyl * 64 + kf];
        bk1.u[p] = *(const uint32_t*)&sK[cur][keyl * 64 + 32 + kf];
      }
      v8f z = {};
      v8f s0 = __builtin_amdgcn_wmma_f32_16x16x32_bf16(
          false, aq[0].v, false, bk0.v, (short)0, z, false, false);
      s[t].v = __builtin_amdgcn_wmma_f32_16x16x32_bf16(
          false, aq[1].v, false, bk1.v, (short)0, s0, false, false);
    }

    // online softmax; row i+8*half lives in element i across lanes of this half
    for (int i = 0; i < 8; i++) {
      int row = qbase + i + 8 * half;
      bool pok = row < pfx;
      int key0 = kb + mloc;
      int key1 = kb + 16 + mloc;
      bool a0 = pok || (key0 >= row);
      bool a1 = pok || (key1 >= row);
      float v0 = a0 ? s[0].f[i] : -3e38f;
      float v1 = a1 ? s[1].f[i] : -3e38f;
      float rm = fmaxf(v0, v1);
      for (int msk = 1; msk < 16; msk <<= 1) rm = fmaxf(rm, __shfl_xor(rm, msk, 32));
      float mnew  = fmaxf(mrow[i], rm);
      float scale = __expf(mrow[i] - mnew);
      mrow[i] = mnew;
      float p0 = a0 ? __expf(v0 - mnew) : 0.f;
      float p1 = a1 ? __expf(v1 - mnew) : 0.f;
      float rs = p0 + p1;
      for (int msk = 1; msk < 16; msk <<= 1) rs += __shfl_xor(rs, msk, 32);
      lrow[i] = lrow[i] * scale + rs;
      sP[(i + 8 * half) * 32 + mloc]      = f2bf(p0);
      sP[(i + 8 * half) * 32 + 16 + mloc] = f2bf(p1);
      for (int j = 0; j < 4; j++) o[j].f[i] *= scale;
    }
    __syncthreads();

    // P as A fragment; V as 4 B fragments; accumulate O
    Frag16 ap;
    for (int p = 0; p < 8; p++) {
      int k = 2 * (p & 3) + 16 * (p >> 2) + 8 * half;
      ap.u[p] = *(const uint32_t*)&sP[mloc * 32 + k];
    }
    for (int j = 0; j < 4; j++) {
      Frag16 bv;
      int col = 16 * j + mloc;
      for (int p = 0; p < 8; p++) {
        int k = 2 * (p & 3) + 16 * (p >> 2) + 8 * half;
        bv.e[2 * p]     = sV[cur][k * 64 + col];
        bv.e[2 * p + 1] = sV[cur][(k + 1) * 64 + col];
      }
      o[j].v = __builtin_amdgcn_wmma_f32_16x16x32_bf16(
          false, ap.v, false, bv.v, (short)0, o[j].v, false, false);
    }
    __syncthreads();
  }

  for (int i = 0; i < 8; i++) {
    float inv = 1.f / lrow[i];
    int row = qbase + i + 8 * half;
    for (int j = 0; j < 4; j++)
      O[(size_t)(bz * SS + row) * DM + headoff + 16 * j + mloc] = f2bf(o[j].f[i] * inv);
  }
}

// ---------------- residual + LayerNorm ----------------
// out = g * (x+d - mean) / (std + eps) + b, biased std (matches reference)
template<int WBF16>
__global__ __launch_bounds__(128)
void ln_kernel(const float* __restrict__ x, const float* __restrict__ d,
               const float* __restrict__ g, const float* __restrict__ b,
               float* __restrict__ outf, uint16_t* __restrict__ outb) {
  __shared__ float red[128];
  const int row = blockIdx.x;
  const int tid = threadIdx.x;
  const size_t base = (size_t)row * DM;
  float v[4];
  for (int i = 0; i < 4; i++) {
    int c = tid + i * 128;
    v[i] = x[base + c] + d[base + c];
  }
  float s = v[0] + v[1] + v[2] + v[3];
  red[tid] = s; __syncthreads();
  for (int off = 64; off > 0; off >>= 1) {
    if (tid < off) red[tid] += red[tid + off];
    __syncthreads();
  }
  float mean = red[0] * (1.0f / DM);
  __syncthreads();
  float sq = 0.f;
  for (int i = 0; i < 4; i++) { float t = v[i] - mean; sq += t * t; }
  red[tid] = sq; __syncthreads();
  for (int off = 64; off > 0; off >>= 1) {
    if (tid < off) red[tid] += red[tid + off];
    __syncthreads();
  }
  float var = red[0] * (1.0f / DM);
  float inv = 1.f / (sqrtf(var) + EPSV);
  for (int i = 0; i < 4; i++) {
    int c = tid + i * 128;
    float y = g[c] * (v[i] - mean) * inv + b[c];
    outf[base + c] = y;
    if (WBF16) outb[base + c] = f2bf(y);
  }
}

// ---------------- launcher ----------------
extern "C" void kernel_launch(void* const* d_in, const int* in_sizes, int n_in,
                              void* d_out, int out_size, void* d_ws, size_t ws_size,
                              hipStream_t stream) {
  (void)in_sizes; (void)n_in; (void)out_size; (void)ws_size;
  const float* x      = (const float*)d_in[0];
  const int*   prefix = (const int*)d_in[1];
  const float* Wq  = (const float*)d_in[2];
  const float* bq  = (const float*)d_in[3];
  const float* Wk  = (const float*)d_in[4];
  const float* bk  = (const float*)d_in[5];
  const float* Wv  = (const float*)d_in[6];
  const float* bv  = (const float*)d_in[7];
  const float* Wo  = (const float*)d_in[8];
  const float* bo  = (const float*)d_in[9];
  const float* g1  = (const float*)d_in[10];
  const float* b1  = (const float*)d_in[11];
  const float* W1  = (const float*)d_in[12];
  const float* b1f = (const float*)d_in[13];
  const float* W2  = (const float*)d_in[14];
  const float* b2f = (const float*)d_in[15];
  const float* g2  = (const float*)d_in[16];
  const float* b2  = (const float*)d_in[17];

  char* ws = (char*)d_ws;
  size_t off = 0;
  auto alloc = [&](size_t bytes) -> void* {
    void* p = ws + off;
    off = (off + bytes + 255) & ~(size_t)255;
    return p;
  };

  const int M = BB * SS;  // 8192 rows
  uint16_t* xb    = (uint16_t*)alloc((size_t)M * DM * 2);
  uint16_t* Wqb   = (uint16_t*)alloc((size_t)DM * DM * 2);
  uint16_t* Wkb   = (uint16_t*)alloc((size_t)DM * DM * 2);
  uint16_t* Wvb   = (uint16_t*)alloc((size_t)DM * DM * 2);
  uint16_t* Wob   = (uint16_t*)alloc((size_t)DM * DM * 2);
  uint16_t* W1b   = (uint16_t*)alloc((size_t)DM * DFF * 2);
  uint16_t* W2b   = (uint16_t*)alloc((size_t)DFF * DM * 2);
  uint16_t* Qb    = (uint16_t*)alloc((size_t)M * DM * 2);
  uint16_t* Kb    = (uint16_t*)alloc((size_t)M * DM * 2);
  uint16_t* Vb    = (uint16_t*)alloc((size_t)M * DM * 2);
  uint16_t* attnb = (uint16_t*)alloc((size_t)M * DM * 2);
  float*    tmp   = (float*)alloc((size_t)M * DM * 4);
  float*    hf    = (float*)alloc((size_t)M * DM * 4);
  uint16_t* hb    = (uint16_t*)alloc((size_t)M * DM * 2);
  uint16_t* ffb   = (uint16_t*)alloc((size_t)M * DFF * 2);

  // 1) converts
  cvt_f32_bf16_kernel<<<512, 256, 0, stream>>>(x,  xb,  M * DM);
  cvt_f32_bf16_kernel<<<256, 256, 0, stream>>>(Wq, Wqb, DM * DM);
  cvt_f32_bf16_kernel<<<256, 256, 0, stream>>>(Wk, Wkb, DM * DM);
  cvt_f32_bf16_kernel<<<256, 256, 0, stream>>>(Wv, Wvb, DM * DM);
  cvt_f32_bf16_kernel<<<256, 256, 0, stream>>>(Wo, Wob, DM * DM);
  cvt_f32_bf16_kernel<<<512, 256, 0, stream>>>(W1, W1b, DM * DFF);
  cvt_f32_bf16_kernel<<<512, 256, 0, stream>>>(W2, W2b, DFF * DM);

  // 2) Q/K/V projections (bf16 out)
  dim3 gq(DM / 64, M / 128);
  gemm_bf16_kernel<0, 0, 1><<<gq, 256, 0, stream>>>(xb, Wqb, bq, nullptr, Qb, DM, DM);
  gemm_bf16_kernel<0, 0, 1><<<gq, 256, 0, stream>>>(xb, Wkb, bk, nullptr, Kb, DM, DM);
  gemm_bf16_kernel<0, 0, 1><<<gq, 256, 0, stream>>>(xb, Wvb, bv, nullptr, Vb, DM, DM);

  // 3) attention
  attn_kernel<<<dim3(SS / 16, HH, BB), 32, 0, stream>>>(Qb, Kb, Vb, prefix, attnb);

  // 4) output projection (f32 out)
  gemm_bf16_kernel<0, 1, 0><<<gq, 256, 0, stream>>>(attnb, Wob, bo, tmp, nullptr, DM, DM);

  // 5) h = LN(x + attn)  (f32 + bf16)
  ln_kernel<1><<<M, 128, 0, stream>>>(x, tmp, g1, b1, hf, hb);

  // 6) FFN: relu(h @ W1 + b1f) -> bf16
  gemm_bf16_kernel<1, 0, 1><<<dim3(DFF / 64, M / 128), 256, 0, stream>>>(hb, W1b, b1f, nullptr, ffb, DFF, DM);

  // 7) ff @ W2 + b2f -> f32
  gemm_bf16_kernel<0, 1, 0><<<gq, 256, 0, stream>>>(ffb, W2b, b2f, tmp, nullptr, DM, DFF);

  // 8) out = LN(h + ff)
  ln_kernel<0><<<M, 128, 0, stream>>>(hf, tmp, g2, b2, (float*)d_out, nullptr);
}
